// LNNStep_20993800143482
// MI455X (gfx1250) — matrix operand order repaired and checked by reading, not compiled, not run on hardware
//
#include <hip/hip_runtime.h>
#include <stdint.h>

typedef __attribute__((ext_vector_type(16))) __bf16 v16bf;
typedef __attribute__((ext_vector_type(8)))  float  v8f;

constexpr int   kB    = 8192;
constexpr int   kIn   = 2048;
constexpr int   kR    = 4096;
constexpr int   kK    = kIn + kR;   // 6144, fused K
constexpr float kLeak = 0.3f;

constexpr int TM   = 128;   // batch rows per workgroup
constexpr int TN   = 128;   // reservoir cols per workgroup
constexpr int TK   = 32;    // k per WMMA step (bf16)
constexpr int LDSW = 20;    // dwords per LDS tile row: 16 data (32 bf16) + 4 pad
constexpr int NKT  = kK / TK;  // 192 k-tiles
constexpr int PFD  = 8;     // L2 prefetch distance in k-tiles

// fp32 -> bf16 (round-to-nearest), pack two into one dword with a single
// v_perm_b32: result = { bf16(b), bf16(a) }  (a in low half = even k)
__device__ __forceinline__ uint32_t pack2_bf16(float a, float b) {
  uint32_t ua = __float_as_uint(a) + 0x8000u;
  uint32_t ub = __float_as_uint(b) + 0x8000u;
  return __builtin_amdgcn_perm(ub, ua, 0x07060302u);
}

// tanh via TRANS-pipe exp: ~6 VALU/TRANS ops vs ~25 for libm tanhf
__device__ __forceinline__ float tanh_fast(float x) {
  float ax = __builtin_fabsf(x);
  float e  = __expf(-2.0f * ax);
  float t  = __fdividef(1.0f - e, 1.0f + e);
  return __builtin_copysignf(t, x);
}

union FragCast { uint4 q[2]; v16bf v; };

__global__ __launch_bounds__(256, 2)
void esn_step_wmma(const float* __restrict__ X,    // [8192, 2048] inputs
                   const float* __restrict__ S,    // [8192, 4096] prev_state
                   const float* __restrict__ Wi,   // [4096, 2048]
                   const float* __restrict__ Wr,   // [4096, 4096]
                   float* __restrict__ O)          // [8192, 4096] new state
{
  // ping-pong double buffers: one __syncthreads per k-tile
  __shared__ uint32_t lA[2][TM * LDSW];
  __shared__ uint32_t lB[2][TN * LDSW];

  const int tid  = threadIdx.x;
  const int lane = tid & 31;
  const int wave = tid >> 5;
  const int wm   = wave >> 2;          // 0..1 -> 64-row slab
  const int wn   = wave & 3;           // 0..3 -> 32-col slab

  const int n0 = blockIdx.x * TN;      // reservoir col base
  const int m0 = blockIdx.y * TM;      // batch row base

  // staging: 2 threads per tile row, each covers 16 consecutive k (fp32)
  const int sRow = tid >> 1;           // 0..127
  const int sK   = (tid & 1) * 16;     // 0 or 16

  float aReg[16];
  float bReg[16];

  v8f acc[4][2];
#pragma unroll
  for (int i = 0; i < 4; ++i)
#pragma unroll
    for (int j = 0; j < 2; ++j)
      acc[i][j] = v8f{};

  auto srcs = [&](int kBase, const float*& aSrc, const float*& bSrc) {
    if (kBase < kIn) {                 // first GEMM half: inputs @ Wi^T
      aSrc = X  + (size_t)(m0 + sRow) * kIn + (kBase + sK);
      bSrc = Wi + (size_t)(n0 + sRow) * kIn + (kBase + sK);
    } else {                           // second half: prev_state @ Wr^T
      const int kb = kBase - kIn;
      aSrc = S  + (size_t)(m0 + sRow) * kR + (kb + sK);
      bSrc = Wr + (size_t)(n0 + sRow) * kR + (kb + sK);
    }
  };

  auto stage_load = [&](int kt) {
    const float* aSrc;
    const float* bSrc;
    srcs(kt * TK, aSrc, bSrc);
#pragma unroll
    for (int i = 0; i < 4; ++i) {
      float4 va = ((const float4*)aSrc)[i];
      float4 vb = ((const float4*)bSrc)[i];
      aReg[4*i+0] = va.x; aReg[4*i+1] = va.y; aReg[4*i+2] = va.z; aReg[4*i+3] = va.w;
      bReg[4*i+0] = vb.x; bReg[4*i+1] = vb.y; bReg[4*i+2] = vb.z; bReg[4*i+3] = vb.w;
    }
    // warm L2 ahead of the stream (global_prefetch_b8)
    const int kpf = (kt + PFD) * TK;
    if (kpf < kK) {
      const float* aP;
      const float* bP;
      srcs(kpf, aP, bP);
      __builtin_prefetch(aP, 0, 1);
      __builtin_prefetch(bP, 0, 1);
    }
  };

  auto stage_store = [&](int buf) {
    uint32_t pa[8], pb[8];
#pragma unroll
    for (int i = 0; i < 8; ++i) {
      pa[i] = pack2_bf16(aReg[2*i], aReg[2*i+1]);
      pb[i] = pack2_bf16(bReg[2*i], bReg[2*i+1]);
    }
    uint32_t* da = &lA[buf][sRow * LDSW + (sK >> 1)];
    uint32_t* db = &lB[buf][sRow * LDSW + (sK >> 1)];
    ((uint4*)da)[0] = uint4{pa[0], pa[1], pa[2], pa[3]};
    ((uint4*)da)[1] = uint4{pa[4], pa[5], pa[6], pa[7]};
    ((uint4*)db)[0] = uint4{pb[0], pb[1], pb[2], pb[3]};
    ((uint4*)db)[1] = uint4{pb[4], pb[5], pb[6], pb[7]};
  };

  const int mrow  = lane & 15;
  const int khalf = lane >> 4;

  auto compute = [&](int buf) {
    v16bf afrag[4];
    v16bf bfrag[2];
    // A fragment (16x32 bf16): lane holds row lane%16;
    // VGPR0-7 <- kpairs khalf*8+0..7
#pragma unroll
    for (int mt = 0; mt < 4; ++mt) {
      const uint32_t* p = &lA[buf][(wm*64 + mt*16 + mrow) * LDSW];
      FragCast f;
      f.q[0] = *(const uint4*)(p + khalf*8);
      f.q[1] = *(const uint4*)(p + khalf*8 + 4);
      afrag[mt] = f.v;
    }
    // B fragment (32x16 bf16): lane holds column lane%16;
    // lanes 0-15 kpairs 0..7, lanes 16-31 kpairs 8..15
#pragma unroll
    for (int nt = 0; nt < 2; ++nt) {
      const uint32_t* p = &lB[buf][(wn*32 + nt*16 + mrow) * LDSW + khalf*8];
      FragCast f;
      f.q[0] = *(const uint4*)(p);
      f.q[1] = *(const uint4*)(p + 4);
      bfrag[nt] = f.v;
    }
#pragma unroll
    for (int mt = 0; mt < 4; ++mt)
#pragma unroll
      for (int nt = 0; nt < 2; ++nt)
        acc[mt][nt] = __builtin_amdgcn_wmma_f32_16x16x32_bf16(
            false, afrag[mt], false, bfrag[nt],
            (short)0, acc[mt][nt], false, false);
  };

  // software pipeline, one barrier per k-tile:
  //   iter kt: fetch regs for kt+1 | barrier | store buf^1 for kt+1 | WMMA on buf
  stage_load(0);
  stage_store(0);
  for (int kt = 0; kt < NKT; ++kt) {
    const int cur = kt & 1;
    if (kt + 1 < NKT) stage_load(kt + 1);
    __syncthreads();
    if (kt + 1 < NKT) stage_store(cur ^ 1);
    compute(cur);
  }

  // epilogue: out = (1-leak)*prev + leak*tanh(acc)
  // C/D layout: VGPR v -> row (v + 8*khalf), lane%16 -> col
#pragma unroll
  for (int mt = 0; mt < 4; ++mt) {
#pragma unroll
    for (int nt = 0; nt < 2; ++nt) {
      const int mBase = m0 + wm*64 + mt*16 + khalf*8;
      const int nCol  = n0 + wn*32 + nt*16 + mrow;
#pragma unroll
      for (int v = 0; v < 8; ++v) {
        const size_t idx = (size_t)(mBase + v) * kR + nCol;
        const float ps = S[idx];
        O[idx] = (1.0f - kLeak) * ps + kLeak * tanh_fast(acc[mt][nt][v]);
      }
    }
  }
}

extern "C" void kernel_launch(void* const* d_in, const int* in_sizes, int n_in,
                              void* d_out, int out_size, void* d_ws, size_t ws_size,
                              hipStream_t stream) {
  const float* inputs = (const float*)d_in[0];   // [8192, 2048]
  const float* prev   = (const float*)d_in[1];   // [8192, 4096]
  const float* Wi     = (const float*)d_in[2];   // [4096, 2048]
  const float* Wr     = (const float*)d_in[3];   // [4096, 4096]
  float* out = (float*)d_out;                    // [8192, 4096]

  dim3 grid(kR / TN, kB / TM);   // 32 x 64 workgroups
  esn_step_wmma<<<grid, 256, 0, stream>>>(inputs, prev, Wi, Wr, out);
}
